// CrossScaleGNN_52656299049428
// MI455X (gfx1250) — matrix operand order repaired and checked by reading, not compiled, not run on hardware
//
#include <hip/hip_runtime.h>
#include <hip/hip_bf16.h>
#include <cmath>

typedef __attribute__((ext_vector_type(16))) _Float16 v16h;
typedef __attribute__((ext_vector_type(8)))  float    v8f;

#define DD 512
#define SS 6
#define NBATCH_TILE 16
#define MROWS 96            // 16 batches * 6 scales
#define ABLK 40             // 32 data halves + 8 pad: 80-byte row stride, 16B aligned
#define KBLKS 16            // 512 / 32
#define ABUF_HALFS (KBLKS * MROWS * ABLK)   // 61440 halves = 122880 B
#define MPITCH 520          // row-major pitch (halves) for the m buffer (reuses A region)

// ---------------- prep kernels ----------------

// adj = softmax(scale_emb @ scale_emb.T)   (6x6)
__global__ void prep_adj(const float* __restrict__ se, float* __restrict__ adj) {
  int i = threadIdx.x;
  if (i < SS) {
    float logits[SS];
    float mx = -1e30f;
    for (int j = 0; j < SS; ++j) {
      float acc = 0.f;
      for (int k = 0; k < DD; ++k) acc += se[i * DD + k] * se[j * DD + k];
      logits[j] = acc;
      mx = fmaxf(mx, acc);
    }
    float s = 0.f;
    for (int j = 0; j < SS; ++j) { logits[j] = expf(logits[j] - mx); s += logits[j]; }
    float inv = 1.0f / s;
    for (int j = 0; j < SS; ++j) adj[i * SS + j] = logits[j] * inv;
  }
}

// f32 -> f16 weight conversion (both weight matrices)
__global__ void prep_convert(const float* __restrict__ W1, const float* __restrict__ W2,
                             _Float16* __restrict__ W1h, _Float16* __restrict__ W2h) {
  int idx = blockIdx.x * 256 + threadIdx.x;
  if (idx < DD * DD) {
    W1h[idx] = (_Float16)W1[idx];
    W2h[idx] = (_Float16)W2[idx];
  }
}

// ---------------- fragment helpers ----------------

// A fragment 16x32 f16 from K-blocked LDS: element (row, k) at ((k>>5)*MROWS+row)*ABLK + (k&31).
// ISA packing: lane m holds M=m; halves K = laneHi*8 + {0..7} and 16 + laneHi*8 + {0..7}
// -> two contiguous 16-byte chunks per lane -> 2x ds_load_b128.
__device__ __forceinline__ v16h load_a_frag_blk(const _Float16* __restrict__ base,
                                                int ks, int row, int laneHi) {
  union { uint4 q[2]; v16h v; } f;
  const uint4* p = (const uint4*)(base + ((ks * MROWS + row) * ABLK + laneHi * 8));
  f.q[0] = p[0];      // halves +0  .. +7
  f.q[1] = p[2];      // halves +16 .. +23
  return f.v;
}

// B fragment 32x16 f16 from global (row-major [n][k]); K = laneHi*8 + {0..7} and 16 + laneHi*8 + {0..7}.
// 16 halves per lane in two 8-half chunks -> two 128-bit loads.
__device__ __forceinline__ v16h load_b_frag(const _Float16* __restrict__ W,
                                            int col, int k0, int laneHi) {
  union { uint4 q[2]; v16h v; } f;
  const uint4* p = (const uint4*)(W + col * DD + k0 + laneHi * 8);
  f.q[0] = p[0];
  f.q[1] = p[2];
  return f.v;
}

#define WMMA_F16(af, bf, c) \
  __builtin_amdgcn_wmma_f32_16x16x32_f16(false, (af), false, (bf), (short)0, (c), false, false)

// ---------------- fused main kernel ----------------

__global__ void __launch_bounds__(256, 1)
cross_scale_gnn_fused(const float* __restrict__ h,
                      const _Float16* __restrict__ W1h,
                      const float* __restrict__ b1,
                      const _Float16* __restrict__ W2h,
                      const float* __restrict__ b2,
                      const float* __restrict__ gamma,
                      const float* __restrict__ beta,
                      const float* __restrict__ adj,
                      float* __restrict__ out) {
  extern __shared__ _Float16 smem[];
  _Float16* Abuf = smem;               // K-blocked msg tile; later reused row-major for m
  _Float16* Tbuf = smem + ABUF_HALFS;  // K-blocked hidden tile

  const int tid    = threadIdx.x;
  const int lane   = tid & 31;
  const int wave   = tid >> 5;
  const int laneLo = lane & 15;
  const int laneHi = lane >> 4;

  const long long rowBase = (long long)blockIdx.x * MROWS;  // flattened (b*S+i)

  // ---- Stage A: msg = adj @ h  -> K-blocked LDS f16 tile (d-quads: b128 loads, b64 stores) ----
  {
    float a_[SS * SS];
#pragma unroll
    for (int t = 0; t < SS * SS; ++t) a_[t] = adj[t];

#pragma unroll 1
    for (int idx = tid; idx < NBATCH_TILE * (DD / 4); idx += 256) {
      int bl = idx >> 7;            // / 128 quads per batch-row
      int dq = idx & 127;           // quad index, d = 4*dq
      const float4* hp = (const float4*)(h + (rowBase + (long long)bl * SS) * DD) + dq;
      float4 hv[SS];
#pragma unroll
      for (int j = 0; j < SS; ++j) hv[j] = hp[j * (DD / 4)];
      int d  = dq * 4;
      int eb = (d >> 5) * MROWS * ABLK + (d & 31);
#pragma unroll
      for (int i = 0; i < SS; ++i) {
        float ax = 0.f, ay = 0.f, az = 0.f, aw = 0.f;
#pragma unroll
        for (int j = 0; j < SS; ++j) {
          float a = a_[i * SS + j];
          ax += a * hv[j].x;
          ay += a * hv[j].y;
          az += a * hv[j].z;
          aw += a * hv[j].w;
        }
        union { _Float16 hh[4]; uint2 u; } pk;
        pk.hh[0] = (_Float16)ax;
        pk.hh[1] = (_Float16)ay;
        pk.hh[2] = (_Float16)az;
        pk.hh[3] = (_Float16)aw;
        *(uint2*)(Abuf + eb + (bl * SS + i) * ABLK) = pk.u;
      }
    }
  }
  __syncthreads();

  // ---- Stage B: GEMM1  T = gelu(msg @ W1^T + b1), col-blocks in pairs ----
#pragma unroll 1
  for (int hlf = 0; hlf < 2; ++hlf) {
    const int nb0  = wave + hlf * 16;        // pair: nb0 and nb0+8
    const int col0 = nb0 * 16 + laneLo;
    const int col1 = col0 + 128;
    v8f acc0[6] = {}, acc1[6] = {};
#pragma unroll 1
    for (int ks = 0; ks < KBLKS; ++ks) {
      v16h bf0 = load_b_frag(W1h, col0, ks * 32, laneHi);
      v16h bf1 = load_b_frag(W1h, col1, ks * 32, laneHi);
#pragma unroll
      for (int mb = 0; mb < 6; ++mb) {
        v16h af = load_a_frag_blk(Abuf, ks, mb * 16 + laneLo, laneHi);
        acc0[mb] = WMMA_F16(af, bf0, acc0[mb]);
        acc1[mb] = WMMA_F16(af, bf1, acc1[mb]);
      }
    }
    const float bias0 = b1[col0], bias1 = b1[col1];
    const int tb0 = (col0 >> 5) * MROWS * ABLK + (col0 & 31);
    const int tb1 = (col1 >> 5) * MROWS * ABLK + (col1 & 31);
#pragma unroll
    for (int mb = 0; mb < 6; ++mb) {
#pragma unroll
      for (int r = 0; r < 8; ++r) {
        int row = mb * 16 + r + laneHi * 8;
        float x0 = acc0[mb][r] + bias0;
        float x1 = acc1[mb][r] + bias1;
        Tbuf[tb0 + row * ABLK] =
            (_Float16)(0.5f * x0 * (1.0f + erff(x0 * 0.70710678118654752f)));
        Tbuf[tb1 + row * ABLK] =
            (_Float16)(0.5f * x1 * (1.0f + erff(x1 * 0.70710678118654752f)));
      }
    }
  }
  __syncthreads();

  // ---- Stage C: GEMM2  m = T @ W2^T + b2 -> row-major f16 in recycled A region ----
#pragma unroll 1
  for (int hlf = 0; hlf < 2; ++hlf) {
    const int nb0  = wave + hlf * 16;
    const int col0 = nb0 * 16 + laneLo;
    const int col1 = col0 + 128;
    v8f acc0[6] = {}, acc1[6] = {};
#pragma unroll 1
    for (int ks = 0; ks < KBLKS; ++ks) {
      v16h bf0 = load_b_frag(W2h, col0, ks * 32, laneHi);
      v16h bf1 = load_b_frag(W2h, col1, ks * 32, laneHi);
#pragma unroll
      for (int mb = 0; mb < 6; ++mb) {
        v16h af = load_a_frag_blk(Tbuf, ks, mb * 16 + laneLo, laneHi);
        acc0[mb] = WMMA_F16(af, bf0, acc0[mb]);
        acc1[mb] = WMMA_F16(af, bf1, acc1[mb]);
      }
    }
    const float bias0 = b2[col0], bias1 = b2[col1];
#pragma unroll
    for (int mb = 0; mb < 6; ++mb) {
#pragma unroll
      for (int r = 0; r < 8; ++r) {
        int row = mb * 16 + r + laneHi * 8;
        Abuf[row * MPITCH + col0] = (_Float16)(acc0[mb][r] + bias0);
        Abuf[row * MPITCH + col1] = (_Float16)(acc1[mb][r] + bias1);
      }
    }
  }
  __syncthreads();

  // ---- Stage D: out = LayerNorm(h + m) * gamma + beta, wave-per-row, float4 I/O ----
#pragma unroll 1
  for (int r = wave; r < MROWS; r += 8) {
    const float4*   hp = (const float4*)(h + (rowBase + r) * (long long)DD);
    const _Float16* mp = Abuf + r * MPITCH;
    float4 xs[4];
    float sum = 0.f, sq = 0.f;
#pragma unroll
    for (int ii = 0; ii < 4; ++ii) {
      int d4 = lane + ii * 32;          // float4 index; d = 4*d4
      float4 hv = hp[d4];
      union { uint2 u; _Float16 hx[4]; } mu;
      mu.u = *(const uint2*)(mp + 4 * d4);
      float4 x;
      x.x = hv.x + (float)mu.hx[0];
      x.y = hv.y + (float)mu.hx[1];
      x.z = hv.z + (float)mu.hx[2];
      x.w = hv.w + (float)mu.hx[3];
      xs[ii] = x;
      sum += x.x + x.y + x.z + x.w;
      sq  += x.x * x.x + x.y * x.y + x.z * x.z + x.w * x.w;
    }
#pragma unroll
    for (int off = 16; off > 0; off >>= 1) {
      sum += __shfl_xor(sum, off, 32);
      sq  += __shfl_xor(sq,  off, 32);
    }
    float mean = sum * (1.0f / 512.0f);
    float var  = sq * (1.0f / 512.0f) - mean * mean;
    float rs   = rsqrtf(var + 1e-5f);
    float4* op = (float4*)(out + (rowBase + r) * (long long)DD);
    const float4* gp = (const float4*)gamma;
    const float4* bp = (const float4*)beta;
#pragma unroll
    for (int ii = 0; ii < 4; ++ii) {
      int d4 = lane + ii * 32;
      float4 g = gp[d4], b = bp[d4], x = xs[ii], o;
      o.x = (x.x - mean) * rs * g.x + b.x;
      o.y = (x.y - mean) * rs * g.y + b.y;
      o.z = (x.z - mean) * rs * g.z + b.z;
      o.w = (x.w - mean) * rs * g.w + b.w;
      op[d4] = o;
    }
  }
}

// ---------------- launcher ----------------

extern "C" void kernel_launch(void* const* d_in, const int* in_sizes, int n_in,
                              void* d_out, int out_size, void* d_ws, size_t ws_size,
                              hipStream_t stream) {
  (void)in_sizes; (void)n_in; (void)out_size; (void)ws_size;

  const float* h     = (const float*)d_in[0];
  const float* se    = (const float*)d_in[1];
  const float* W1    = (const float*)d_in[2];
  const float* b1    = (const float*)d_in[3];
  const float* W2    = (const float*)d_in[4];
  const float* b2    = (const float*)d_in[5];
  const float* gamma = (const float*)d_in[6];
  const float* beta  = (const float*)d_in[7];
  float* out = (float*)d_out;

  _Float16* W1h = (_Float16*)d_ws;
  _Float16* W2h = W1h + DD * DD;
  float*    adj = (float*)((char*)d_ws + 2 * (size_t)DD * DD * sizeof(_Float16));

  prep_adj<<<1, 32, 0, stream>>>(se, adj);
  prep_convert<<<(DD * DD + 255) / 256, 256, 0, stream>>>(W1, W2, W1h, W2h);

  const int nblocks = 32768 / NBATCH_TILE;  // 2048
  const size_t smem = (size_t)2 * ABUF_HALFS * sizeof(_Float16);  // ~240 KB
  hipFuncSetAttribute((const void*)cross_scale_gnn_fused,
                      hipFuncAttributeMaxDynamicSharedMemorySize, (int)smem);
  cross_scale_gnn_fused<<<nblocks, 256, smem, stream>>>(h, W1h, b1, W2h, b2,
                                                        gamma, beta, adj, out);
}
